// Mamba3Block_2233382993982
// MI455X (gfx1250) — matrix operand order repaired and checked by reading, not compiled, not run on hardware
//
#include <hip/hip_runtime.h>

// ---------------- problem constants ----------------
#define DM     1024
#define LSEQ   2048
#define BSZ    4
#define HH     16
#define GG     4
#define DPROJ  3084          // H*P*2 + G*N*R*2 + G*3
#define NPAD   3136          // DPROJ padded to multiple of 64
#define TOK    (BSZ*LSEQ)    // 8192 sequence tokens
#define ASTR   32            // LDS tile row stride in bf16 elems (16B-aligned rows for async b128)

typedef __attribute__((ext_vector_type(16))) __bf16 v16bf;
typedef __attribute__((ext_vector_type(8)))  float  v8f;
typedef __attribute__((ext_vector_type(2)))  float  v2f;
typedef unsigned short u16_t;
typedef unsigned int   u32_t;

__device__ inline u16_t f2bf(float f) {
  u32_t x = __float_as_uint(f);
  u32_t r = x + 0x7FFFu + ((x >> 16) & 1u);   // round-to-nearest-even
  return (u16_t)(r >> 16);
}

__device__ inline unsigned lds_off(const void* p) {
  // generic pointer to a __shared__ object: low 32 bits are the LDS byte address
  return (unsigned)(unsigned long long)p;
}
__device__ inline void async_g2l_b128(unsigned loff, const void* g) {
  asm volatile("global_load_async_to_lds_b128 %0, %1, off" :: "v"(loff), "v"(g) : "memory");
}
__device__ inline void async_g2l_b128_o16(unsigned loff, const void* g) {
  asm volatile("global_load_async_to_lds_b128 %0, %1, off offset:16" :: "v"(loff), "v"(g) : "memory");
}
__device__ inline void wait_async0() {
  asm volatile("s_wait_asynccnt 0x0" ::: "memory");
}

// ---------------- fp32 -> bf16 conversion passes ----------------
__global__ void k_cvt_u(const float* __restrict__ u, u16_t* __restrict__ ub) {
  int i = (blockIdx.x * 256 + threadIdx.x) * 4;
#pragma unroll
  for (int q = 0; q < 4; ++q) ub[i + q] = f2bf(u[i + q]);
}

__global__ void k_cvt_w(const float* __restrict__ w, u16_t* __restrict__ wb) {
  int row = blockIdx.x;
  int c   = threadIdx.x * 4;
#pragma unroll
  for (int q = 0; q < 4; ++q)
    wb[(size_t)row * DM + c + q] = (row < DPROJ) ? f2bf(w[(size_t)row * DM + c + q]) : (u16_t)0;
}

// ---------------- WMMA fragment loaders (from LDS bf16 tiles) ----------------
// A-matrix 16x32 bf16 (ISA 7.12.2): lanes 0-15 row M, K={0..7,16..23}; lanes 16-31 K={8..15,24..31}
__device__ inline v16bf ld_fragA(const u16_t* row0, int lane) {
  int m = lane & 15, hi = lane >> 4;
  union { v16bf v; u32_t u[8]; } r;
  const u32_t* p = (const u32_t*)(row0 + m * ASTR);
#pragma unroll
  for (int q = 0; q < 8; ++q) r.u[q] = p[((q & 4) ? 8 : 0) + hi * 4 + (q & 3)];
  return r.v;
}
// B-matrix 32x16 bf16: lane holds column n=lane&15; lanes 0-15 K={0..7,16..23}; lanes 16-31 K={8..15,24..31}
__device__ inline v16bf ld_fragB(const u16_t* row0, int lane) {
  int nn = lane & 15, hi = lane >> 4;
  union { v16bf v; u32_t u[8]; } r;
  const u32_t* p = (const u32_t*)(row0 + nn * ASTR);
#pragma unroll
  for (int q = 0; q < 8; ++q) r.u[q] = p[((q & 4) ? 8 : 0) + hi * 4 + (q & 3)];
  return r.v;
}

// ------- main GEMM: proj = u @ Wt^T + bias (bf16 WMMA, async-to-LDS double buffer) -------
__global__ void __launch_bounds__(256) k_gemm(const u16_t* __restrict__ A,
                                              const u16_t* __restrict__ Wt,
                                              const float* __restrict__ bias,
                                              float* __restrict__ out) {
  __shared__ u16_t sa[2][128 * ASTR];
  __shared__ u16_t sb[2][64 * ASTR];
  int tid  = threadIdx.x;
  int m0   = blockIdx.x * 128, n0 = blockIdx.y * 64;
  int wave = tid >> 5, lane = tid & 31;
  int mr   = wave >> 1, nc = wave & 1;      // 4x2 wave grid -> 32x32 per wave
  int row  = tid >> 1, half = tid & 1;
  v8f acc00 = {}, acc01 = {}, acc10 = {}, acc11 = {};

  auto stage = [&](int kblk, int bufi) {
    unsigned la = lds_off(&sa[bufi][row * ASTR + half * 16]);
    const char* ga = (const char*)(A + (size_t)(m0 + row) * DM + kblk + half * 16);
    async_g2l_b128(la, ga);
    async_g2l_b128_o16(la, ga);
    if (tid < 128) {   // B tile: 64 rows x 32 k
      unsigned lb = lds_off(&sb[bufi][row * ASTR + half * 16]);
      const char* gb = (const char*)(Wt + (size_t)(n0 + row) * DM + kblk + half * 16);
      async_g2l_b128(lb, gb);
      async_g2l_b128_o16(lb, gb);
    }
  };

  stage(0, 0);
  wait_async0();
  __syncthreads();

  for (int it = 0; it < DM / 32; ++it) {
    int bufi = it & 1;
    if (it + 1 < DM / 32) stage((it + 1) * 32, bufi ^ 1);   // overlap DMA with math
    v16bf a0 = ld_fragA(&sa[bufi][(mr * 32) * ASTR], lane);
    v16bf a1 = ld_fragA(&sa[bufi][(mr * 32 + 16) * ASTR], lane);
    v16bf b0 = ld_fragB(&sb[bufi][(nc * 32) * ASTR], lane);
    v16bf b1 = ld_fragB(&sb[bufi][(nc * 32 + 16) * ASTR], lane);
    acc00 = __builtin_amdgcn_wmma_f32_16x16x32_bf16(false, a0, false, b0, (short)0, acc00, false, false);
    acc01 = __builtin_amdgcn_wmma_f32_16x16x32_bf16(false, a0, false, b1, (short)0, acc01, false, false);
    acc10 = __builtin_amdgcn_wmma_f32_16x16x32_bf16(false, a1, false, b0, (short)0, acc10, false, false);
    acc11 = __builtin_amdgcn_wmma_f32_16x16x32_bf16(false, a1, false, b1, (short)0, acc11, false, false);
    wait_async0();
    __syncthreads();
  }

  // epilogue: C/D layout lane -> (M = q + 8*hi, N = lane&15)
  int hi = lane >> 4, nl = lane & 15;
#pragma unroll
  for (int q = 0; q < 8; ++q) {
    int Mb = m0 + mr * 32 + q + 8 * hi;
    int Nb = n0 + nc * 32 + nl;
    int N1 = Nb + 16;
    float bq0 = (Nb < DPROJ) ? bias[Nb] : 0.f;
    float bq1 = (N1 < DPROJ) ? bias[N1] : 0.f;
    out[(size_t)Mb * NPAD + Nb]        = acc00[q] + bq0;
    out[(size_t)Mb * NPAD + N1]        = acc01[q] + bq1;
    out[(size_t)(Mb + 16) * NPAD + Nb] = acc10[q] + bq0;
    out[(size_t)(Mb + 16) * NPAD + N1] = acc11[q] + bq1;
  }
}

// ---------------- dt / alpha / cumulative-dt (sequential per (b,g)) ----------------
__global__ void k_seq(const float* __restrict__ proj, float* __restrict__ dt,
                      float* __restrict__ al, float* __restrict__ cdt) {
  int id = threadIdx.x;
  if (id >= BSZ * GG) return;
  int b = id >> 2, g = id & 3;
  float c = 0.f;
  for (int t = 0; t < LSEQ; ++t) {
    size_t base = (size_t)(b * LSEQ + t) * NPAD;
    float draw = proj[base + 3072 + g];
    float dtt  = (draw > 20.f) ? draw : log1pf(expf(draw));      // softplus
    float a    = expf(dtt * (-expf(proj[base + 3076 + g])));     // exp(dt * A)
    c += dtt;
    int o = (b * LSEQ + t) * GG + g;
    dt[o] = dtt; al[o] = a; cdt[o] = c;
  }
}

// ---------------- RMS-norm + bias + rotary rotation of B/C ----------------
__global__ void __launch_bounds__(128) k_bc(const float* __restrict__ proj,
    const float* __restrict__ normB, const float* __restrict__ normC,
    const float* __restrict__ biasB, const float* __restrict__ biasC,
    const float* __restrict__ thlog, const float* __restrict__ cdt,
    float* __restrict__ Brot, float* __restrict__ Crot) {
  int tok = blockIdx.x;
  int g = threadIdx.x >> 5, lane = threadIdx.x & 31;   // one wave per group
  size_t pb = (size_t)tok * NPAD;
  float ang = cdt[tok * GG + g] * expf(thlog[g * 32 + lane]);
  float cs = cosf(ang), sn = sinf(ang);
#pragma unroll
  for (int which = 0; which < 2; ++which) {
    const float* src = proj + pb + (which ? 2560 : 2048) + g * 128 + lane * 4;
    float v0 = src[0], v1 = src[1], v2 = src[2], v3 = src[3];
    float ss = v0 * v0 + v1 * v1 + v2 * v2 + v3 * v3;
    for (int off = 16; off; off >>= 1) ss += __shfl_xor(ss, off, 32);
    float sc = rsqrtf(ss / 128.f + 1e-5f);
    const float* nw = (which ? normC : normB) + lane * 4;
    const float* bb = (which ? biasC : biasB) + g * 128 + lane * 4;
    v0 = v0 * sc * nw[0] + bb[0]; v1 = v1 * sc * nw[1] + bb[1];
    v2 = v2 * sc * nw[2] + bb[2]; v3 = v3 * sc * nw[3] + bb[3];
    float o0 = v0 * cs - v2 * sn, o2 = v0 * sn + v2 * cs;   // r=0
    float o1 = v1 * cs - v3 * sn, o3 = v1 * sn + v3 * cs;   // r=1
    float* dst = (which ? Crot : Brot) + ((size_t)tok * GG + g) * 128 + lane * 4;
    dst[0] = o0; dst[1] = o1; dst[2] = o2; dst[3] = o3;
  }
}

// ---------------- xup kron-MoE: (8x8) -> (8x16) top-2 ----------------
__global__ void __launch_bounds__(256) k_xup(const float* __restrict__ proj,
    const float* __restrict__ router, const float* __restrict__ Ae,
    const float* __restrict__ Be, const float* __restrict__ scale,
    const float* __restrict__ bias, float* __restrict__ xout) {
  __shared__ float xsh[8][64];
  __shared__ float lg[8][64];
  __shared__ float meta[8][4];
  int wave = threadIdx.x >> 5, lane = threadIdx.x & 31;
  int tok = blockIdx.x * 8 + wave;          // (b,l,h) flat
  int bl = tok >> 4, h = tok & 15;
  const float* xp = proj + (size_t)bl * NPAD + 1024 + h * 64;
  xsh[wave][lane * 2]     = xp[lane * 2];
  xsh[wave][lane * 2 + 1] = xp[lane * 2 + 1];
  __syncthreads();
  float l0 = 0.f, l1 = 0.f;
  const float* r0 = router + lane * 64;
  const float* r1 = router + (lane + 32) * 64;
  for (int q = 0; q < 64; ++q) { float xv = xsh[wave][q]; l0 += r0[q] * xv; l1 += r1[q] * xv; }
  lg[wave][lane] = l0; lg[wave][lane + 32] = l1;
  __syncthreads();
  if (lane == 0) {
    int i0 = 0; float v0 = lg[wave][0];
    for (int q = 1; q < 64; ++q) if (lg[wave][q] > v0) { v0 = lg[wave][q]; i0 = q; }
    int i1 = 0; float v1 = -3.4e38f;
    for (int q = 0; q < 64; ++q) if (q != i0 && lg[wave][q] > v1) { v1 = lg[wave][q]; i1 = q; }
    float e1 = expf(v1 - v0);
    meta[wave][0] = 1.f / (1.f + e1); meta[wave][1] = e1 / (1.f + e1);
    meta[wave][2] = __int_as_float(i0); meta[wave][3] = __int_as_float(i1);
  }
  __syncthreads();
  float pk[2] = { meta[wave][0], meta[wave][1] };
  int   ek[2] = { __float_as_int(meta[wave][2]), __float_as_int(meta[wave][3]) };
  float sc = scale[0];
#pragma unroll
  for (int q = 0; q < 4; ++q) {
    int f = lane * 4 + q, o = f >> 4, p = f & 15;
    float acc = 0.f;
    for (int k = 0; k < 2; ++k) {
      const float* Ax = Ae + ek[k] * 64 + o * 8;
      const float* Bx = Be + ek[k] * 128 + p * 8;
      float y = 0.f;
      for (int i = 0; i < 8; ++i) {
        float t = 0.f;
        const float* xr = &xsh[wave][i * 8];
        for (int j = 0; j < 8; ++j) t += xr[j] * Bx[j];
        y += Ax[i] * t;
      }
      acc += pk[k] * y;
    }
    xout[(size_t)tok * 128 + f] = acc * sc + bias[f];
  }
}

// ------- sequential SSM scan: state in WMMA accumulators, rank-2 update via f32 WMMA -------
// one block = 4 waves per (b,h); wave w owns n-slice [16w,16w+16) as 4 v8f tiles (p tiles of 16)
__global__ void __launch_bounds__(128) k_scan(const float* __restrict__ Brot,
    const float* __restrict__ Crot, const float* __restrict__ xup,
    const float* __restrict__ dtb, const float* __restrict__ alb,
    float* __restrict__ ys) {
  __shared__ float sB[128], sC[128], sx[128];
  __shared__ float yp[4][128];
  int b = blockIdx.x >> 4, h = blockIdx.x & 15, g = h >> 2;
  int tid = threadIdx.x;
  int w = tid >> 5, lane = tid & 31;
  int m = lane & 15, hi = lane >> 4;
  v8f h0 = {}, h1 = {}, h2 = {}, h3 = {};

  for (int t = 0; t < LSEQ; ++t) {
    size_t tg = ((size_t)(b * LSEQ + t) * GG + g) * 128;
    size_t th = ((size_t)(b * LSEQ + t) * HH + h) * 128;
    sB[tid] = Brot[tg + tid];
    sC[tid] = Crot[tg + tid];
    sx[tid] = xup[th + tid];
    float dtt = dtb[(b * LSEQ + t) * GG + g];
    float al  = alb[(b * LSEQ + t) * GG + g];
    __syncthreads();

    // A frag (16x4 f32): row = local n = m, K0/K1 = dt*Br[n,0..1] (lanes 0-15), K2/K3 = 0
    v2f af;
    af.x = hi ? 0.f : dtt * sB[(16 * w + m) * 2 + 0];
    af.y = hi ? 0.f : dtt * sB[(16 * w + m) * 2 + 1];
    // pre-scale state by alpha, then h += (dt*Br) x^T on the matrix pipe
#pragma unroll
    for (int q = 0; q < 8; ++q) { h0[q] *= al; h1[q] *= al; h2[q] *= al; h3[q] *= al; }
    v2f bf;
    bf.x = hi ? 0.f : sx[(0 * 16 + m) * 2 + 0];
    bf.y = hi ? 0.f : sx[(0 * 16 + m) * 2 + 1];
    h0 = __builtin_amdgcn_wmma_f32_16x16x4_f32(false, af, false, bf, (short)0, h0, false, false);
    bf.x = hi ? 0.f : sx[(1 * 16 + m) * 2 + 0];
    bf.y = hi ? 0.f : sx[(1 * 16 + m) * 2 + 1];
    h1 = __builtin_amdgcn_wmma_f32_16x16x4_f32(false, af, false, bf, (short)0, h1, false, false);
    bf.x = hi ? 0.f : sx[(2 * 16 + m) * 2 + 0];
    bf.y = hi ? 0.f : sx[(2 * 16 + m) * 2 + 1];
    h2 = __builtin_amdgcn_wmma_f32_16x16x4_f32(false, af, false, bf, (short)0, h2, false, false);
    bf.x = hi ? 0.f : sx[(3 * 16 + m) * 2 + 0];
    bf.y = hi ? 0.f : sx[(3 * 16 + m) * 2 + 1];
    h3 = __builtin_amdgcn_wmma_f32_16x16x4_f32(false, af, false, bf, (short)0, h3, false, false);

    // y[p,r] = sum_n h[n,p]*C[n,r]; lane holds h[n=16w+q+8hi][p=16pt+m]
#pragma unroll
    for (int pt = 0; pt < 4; ++pt) {
      const v8f& hp = pt == 0 ? h0 : pt == 1 ? h1 : pt == 2 ? h2 : h3;
      float s0 = 0.f, s1 = 0.f;
#pragma unroll
      for (int q = 0; q < 8; ++q) {
        int n = 16 * w + q + 8 * hi;
        float hv = hp[q];
        s0 += hv * sC[n * 2 + 0];
        s1 += hv * sC[n * 2 + 1];
      }
      s0 += __shfl_xor(s0, 16, 32);
      s1 += __shfl_xor(s1, 16, 32);
      if (hi == 0) {
        yp[w][(16 * pt + m) * 2 + 0] = s0;
        yp[w][(16 * pt + m) * 2 + 1] = s1;
      }
    }
    __syncthreads();
    ys[th + tid] = yp[0][tid] + yp[1][tid] + yp[2][tid] + yp[3][tid];
    __syncthreads();
  }
}

// ---------------- ydown @ + skip*D, RMS, pregate, sigmoid(z) gate ----------------
__global__ void __launch_bounds__(256) k_ydown(const float* __restrict__ ys,
    const float* __restrict__ ydw, const float* __restrict__ proj,
    const float* __restrict__ Dv, const float* __restrict__ pregate,
    float* __restrict__ yvec) {
  __shared__ float ybuf[2048];
  __shared__ float red[8];
  int tok = blockIdx.x, tid = threadIdx.x;
  for (int q = tid; q < 2048; q += 256) ybuf[q] = ys[(size_t)tok * 2048 + q];
  __syncthreads();
  float vals[4]; float ss = 0.f;
#pragma unroll
  for (int q = 0; q < 4; ++q) {
    int i = tid + 256 * q;
    int h = i >> 6, p = i & 63;
    const float* w = ydw + p * 128;
    const float* yy = ybuf + h * 128;
    float s = 0.f;
    for (int c = 0; c < 128; ++c) s += yy[c] * w[c];
    float v = s + proj[(size_t)tok * NPAD + 1024 + i] * Dv[h];
    vals[q] = v; ss += v * v;
  }
  for (int off = 16; off; off >>= 1) ss += __shfl_xor(ss, off, 32);
  if ((tid & 31) == 0) red[tid >> 5] = ss;
  __syncthreads();
  if (tid == 0) {
    float tot = 0.f;
    for (int q = 0; q < 8; ++q) tot += red[q];
    red[0] = rsqrtf(tot / 1024.f + 1e-5f);
  }
  __syncthreads();
  float sc = red[0];
#pragma unroll
  for (int q = 0; q < 4; ++q) {
    int i = tid + 256 * q;
    float z = proj[(size_t)tok * NPAD + i];
    float sg = 1.f / (1.f + expf(-z));
    yvec[(size_t)tok * 1024 + i] = vals[q] * sc * pregate[i] * sg;
  }
}

// ---------------- oup kron-MoE: (32x32) -> (32x32) top-2 ----------------
__global__ void __launch_bounds__(256) k_oup(const float* __restrict__ yvec,
    const float* __restrict__ router, const float* __restrict__ Ae,
    const float* __restrict__ Be, const float* __restrict__ scale,
    const float* __restrict__ bias, float* __restrict__ outp) {
  __shared__ float X[1024];
  __shared__ float T[1024];
  __shared__ float lg[64];
  __shared__ float meta[4];
  int tok = blockIdx.x, tid = threadIdx.x;
  if (tid < 64) __builtin_prefetch(router + (size_t)tid * 1024, 0, 1);
  for (int q = tid; q < 1024; q += 256) X[q] = yvec[(size_t)tok * 1024 + q];
  __syncthreads();
  if (tid < 64) {
    const float* r = router + (size_t)tid * 1024;
    float s = 0.f;
    for (int q = 0; q < 1024; ++q) s += r[q] * X[q];
    lg[tid] = s;
  }
  __syncthreads();
  if (tid == 0) {
    int i0 = 0; float v0 = lg[0];
    for (int q = 1; q < 64; ++q) if (lg[q] > v0) { v0 = lg[q]; i0 = q; }
    int i1 = 0; float v1 = -3.4e38f;
    for (int q = 0; q < 64; ++q) if (q != i0 && lg[q] > v1) { v1 = lg[q]; i1 = q; }
    float e1 = expf(v1 - v0);
    meta[0] = 1.f / (1.f + e1); meta[1] = e1 / (1.f + e1);
    meta[2] = __int_as_float(i0); meta[3] = __int_as_float(i1);
  }
  __syncthreads();
  float oacc[4] = {0.f, 0.f, 0.f, 0.f};
  for (int k = 0; k < 2; ++k) {
    int e = __float_as_int(meta[2 + k]); float pk = meta[k];
    const float* A  = Ae + (size_t)e * 1024;
    const float* Bm = Be + (size_t)e * 1024;
#pragma unroll
    for (int q = 0; q < 4; ++q) {
      int f = tid + 256 * q, o = f >> 5, j = f & 31;
      float s = 0.f;
      for (int i = 0; i < 32; ++i) s += A[o * 32 + i] * X[i * 32 + j];
      T[f] = s;
    }
    __syncthreads();
#pragma unroll
    for (int q = 0; q < 4; ++q) {
      int f = tid + 256 * q, o = f >> 5, p = f & 31;
      float s = 0.f;
      const float* tr = T + o * 32;
      const float* br = Bm + p * 32;
      for (int j = 0; j < 32; ++j) s += tr[j] * br[j];
      oacc[q] += pk * s;
    }
    __syncthreads();
  }
  float sc = scale[0];
#pragma unroll
  for (int q = 0; q < 4; ++q) {
    int f = tid + 256 * q;
    outp[(size_t)tok * 1024 + f] = oacc[q] * sc + bias[f];
  }
}

// ---------------- launcher ----------------
extern "C" void kernel_launch(void* const* d_in, const int* in_sizes, int n_in,
                              void* d_out, int out_size, void* d_ws, size_t ws_size,
                              hipStream_t stream) {
  (void)in_sizes; (void)n_in; (void)out_size; (void)ws_size;
  const float* u       = (const float*)d_in[0];
  const float* inw     = (const float*)d_in[1];
  const float* inb     = (const float*)d_in[2];
  const float* xrouter = (const float*)d_in[3];
  const float* xA      = (const float*)d_in[4];
  const float* xB      = (const float*)d_in[5];
  const float* xscale  = (const float*)d_in[6];
  const float* xbias   = (const float*)d_in[7];
  const float* ydw     = (const float*)d_in[8];
  const float* orouter = (const float*)d_in[9];
  const float* oA      = (const float*)d_in[10];
  const float* oB      = (const float*)d_in[11];
  const float* oscale  = (const float*)d_in[12];
  const float* obias   = (const float*)d_in[13];
  const float* thlog   = (const float*)d_in[14];
  const float* Dv      = (const float*)d_in[15];
  const float* normB   = (const float*)d_in[16];
  const float* normC   = (const float*)d_in[17];
  const float* biasB   = (const float*)d_in[18];
  const float* biasC   = (const float*)d_in[19];
  const float* pregate = (const float*)d_in[20];
  float* outp = (float*)d_out;

  char* ws = (char*)d_ws;
  size_t off = 0;
  auto take = [&](size_t bytes) { char* p = ws + off; off += (bytes + 255) & ~(size_t)255; return p; };
  u16_t* ub   = (u16_t*)take((size_t)TOK * DM * 2);
  u16_t* wb   = (u16_t*)take((size_t)NPAD * DM * 2);
  float* proj = (float*)take((size_t)TOK * NPAD * 4);
  float* dtb  = (float*)take((size_t)TOK * GG * 4);
  float* alb  = (float*)take((size_t)TOK * GG * 4);
  float* cdt  = (float*)take((size_t)TOK * GG * 4);
  float* Brot = (float*)take((size_t)TOK * GG * 128 * 4);
  float* Crot = (float*)take((size_t)TOK * GG * 128 * 4);
  float* xup  = (float*)take((size_t)TOK * HH * 128 * 4);
  float* ysc  = (float*)take((size_t)TOK * HH * 128 * 4);
  float* yv   = (float*)take((size_t)TOK * DM * 4);

  k_cvt_u<<<(TOK * DM) / 1024, 256, 0, stream>>>(u, ub);
  k_cvt_w<<<NPAD, 256, 0, stream>>>(inw, wb);
  k_gemm<<<dim3(TOK / 128, NPAD / 64), 256, 0, stream>>>(ub, wb, inb, proj);
  k_seq<<<1, 32, 0, stream>>>(proj, dtb, alb, cdt);
  k_bc<<<TOK, 128, 0, stream>>>(proj, normB, normC, biasB, biasC, thlog, cdt, Brot, Crot);
  k_xup<<<(TOK * HH) / 8, 256, 0, stream>>>(proj, xrouter, xA, xB, xscale, xbias, xup);
  k_scan<<<BSZ * HH, 128, 0, stream>>>(Brot, Crot, xup, dtb, alb, ysc);
  k_ydown<<<TOK, 256, 0, stream>>>(ysc, ydw, proj, Dv, pregate, yv);
  k_oup<<<TOK, 256, 0, stream>>>(yv, orouter, oA, oB, oscale, obias, outp);
}
